// AttentionGRU_78185584656776
// MI455X (gfx1250) — matrix operand order repaired and checked by reading, not compiled, not run on hardware
//
#include <hip/hip_runtime.h>
#include <hip/hip_bf16.h>
#include <cstdint>
#include <math.h>

typedef __attribute__((ext_vector_type(16))) _Float16 v16h;
typedef __attribute__((ext_vector_type(8)))  float    v8f;

#define B_SZ   64
#define P_SZ   2048
#define L_SZ   512
#define DM     512
#define NH     8
#define DK     64
#define TILE   64                 // rows of P per workgroup
#define NTILES (P_SZ / TILE)      // 32
#define THREADS 256
#define LN_EPS 1e-5f

// ---- workspace layout (bytes) ----
#define WS_Q    0                 // f32 [B][512]
#define WS_RES  131072            // f32 [B][512]
#define WS_WQ   262144            // f16 [B][16][512]  (head-major, heads 8..15 zero)
#define WS_S    1310720           // f32 [B][8][2048]  raw scores
#define WS_M    5505024           // f32 [B][32][8]    tile max
#define WS_Z    5570560           // f32 [B][32][8]    tile expsum
#define WS_U    5636096           // f32 [B][32][8][512] tile weighted row sums

struct H4 { _Float16 a, b, c, d; };          // 8 bytes
union  VA { int4 q[2]; v16h h; };            // 32 bytes (16 f16)

// ---------------------------------------------------------------------------
// k1: per-batch LN(query) @ W_q  and residual = x_trafic @ res_W + res_b
// ---------------------------------------------------------------------------
__global__ __launch_bounds__(THREADS)
void k1_query_residual(const float* __restrict__ x_trafic,
                       const float* __restrict__ W_q,
                       const float* __restrict__ lnq_g, const float* __restrict__ lnq_b,
                       const float* __restrict__ res_W, const float* __restrict__ res_b,
                       float* __restrict__ Qout, float* __restrict__ Rout) {
  __shared__ float xs[L_SZ];
  __shared__ float qln[L_SZ];
  __shared__ float red[THREADS], red2[THREADS];
  const int b = blockIdx.x, t = threadIdx.x;
  float x0 = x_trafic[b * L_SZ + t];
  float x1 = x_trafic[b * L_SZ + t + 256];
  xs[t] = x0; xs[t + 256] = x1;
  red[t] = x0 + x1; red2[t] = x0 * x0 + x1 * x1;
  __syncthreads();
  for (int s = 128; s > 0; s >>= 1) {
    if (t < s) { red[t] += red[t + s]; red2[t] += red2[t + s]; }
    __syncthreads();
  }
  const float mean = red[0] * (1.0f / L_SZ);
  const float var  = red2[0] * (1.0f / L_SZ) - mean * mean;
  const float rstd = rsqrtf(var + LN_EPS);
  qln[t]       = (x0 - mean) * rstd * lnq_g[t]       + lnq_b[t];
  qln[t + 256] = (x1 - mean) * rstd * lnq_g[t + 256] + lnq_b[t + 256];
  __syncthreads();
  for (int jj = 0; jj < 2; ++jj) {
    const int j = t + jj * 256;
    float q = 0.f, r = 0.f;
    for (int l = 0; l < L_SZ; ++l) {
      q += qln[l] * W_q[(size_t)l * DM + j];
      r += xs[l]  * res_W[(size_t)l * DM + j];
    }
    Qout[(size_t)b * DM + j] = q;
    Rout[(size_t)b * DM + j] = r + res_b[j];
  }
}

// ---------------------------------------------------------------------------
// k2: wq_eff[b,h,l] = (1/8) * sum_d W_k[l, h*64+d] * Q[b, h*64+d]   (f16)
// stored head-major: [b][h(16)][l] so WMMA B-tiles load contiguously
// ---------------------------------------------------------------------------
__global__ __launch_bounds__(THREADS)
void k2_wqeff(const float* __restrict__ W_k, const float* __restrict__ Qin,
              _Float16* __restrict__ wq) {
  __shared__ float Qs[DM];
  const int b = blockIdx.x, t = threadIdx.x;
  Qs[t] = Qin[(size_t)b * DM + t];
  Qs[t + 256] = Qin[(size_t)b * DM + t + 256];
  __syncthreads();
  for (int ll = 0; ll < 2; ++ll) {
    const int l = t + ll * 256;
    const float* wrow = W_k + (size_t)l * DM;
    for (int h = 0; h < NH; ++h) {
      float s = 0.f;
#pragma unroll 8
      for (int d = 0; d < DK; ++d) s += wrow[h * DK + d] * Qs[h * DK + d];
      wq[((size_t)b * 16 + h) * L_SZ + l] = (_Float16)(s * 0.125f); // 1/sqrt(64) folded
    }
    for (int h = NH; h < 16; ++h) wq[((size_t)b * 16 + h) * L_SZ + l] = (_Float16)0.f;
  }
}

// ---------------------------------------------------------------------------
// k3: streaming tile kernel (the HBM-bound pass over x_dynamic, 256 MB total)
//   grid = B * NTILES, 256 threads (8 wave32)
// ---------------------------------------------------------------------------
__global__ __launch_bounds__(THREADS)
void k3_attend(const float* __restrict__ x_dynamic,
               const _Float16* __restrict__ wq,
               const float* __restrict__ lnkv_g, const float* __restrict__ lnkv_b,
               float* __restrict__ Sout, float* __restrict__ Mout,
               float* __restrict__ Zout, float* __restrict__ Uout) {
  extern __shared__ char smem[];
  char*  rows = smem;                                   // TILE*2048 B (f32 -> f16 in place)
  char*  bq   = smem + TILE * 2048;                     // 16 KB  B-matrix (head-major f16)
  float* gbuf = (float*)(smem + TILE * 2048 + 16384);   // 2 KB gamma
  float* bbuf = (float*)(smem + TILE * 2048 + 18432);   // 2 KB beta
  float* scA  = (float*)(smem + TILE * 2048 + 20480);   // [64][8] scores (K-half 0)
  float* scB  = scA + TILE * NH;                        // [64][8] scores (K-half 1)
  float* wbuf = scB + TILE * NH;                        // [64][8] exp weights

  const int t = threadIdx.x;
  const int lane = t & 31, wave = t >> 5;
  const int b = blockIdx.x >> 5;        // NTILES == 32
  const int tile = blockIdx.x & 31;
  const int p0 = tile * TILE;

  // ---- Phase 1: async DMA everything into LDS (ASYNCcnt path) ----
  // x_dynamic is streamed exactly once (zero reuse): non-temporal hint keeps
  // the 256 MB stream from evicting the reused wq/W_v/gamma lines in L2.
  {
    const char* gsrc = (const char*)(x_dynamic + ((size_t)b * P_SZ + p0) * L_SZ);
#pragma unroll 4
    for (int it = 0; it < (TILE * 2048) / (THREADS * 16); ++it) {  // 32 iters
      unsigned off = (unsigned)(it * THREADS * 16 + t * 16);
      unsigned lds = (unsigned)(uintptr_t)(rows + off);
      asm volatile("global_load_async_to_lds_b128 %0, %1, off th:TH_LOAD_NT"
                   :: "v"(lds), "v"(gsrc + off) : "memory");
    }
    const char* gq = (const char*)(wq + (size_t)b * 16 * L_SZ);
#pragma unroll
    for (int it = 0; it < 4; ++it) {                               // 16 KB B-matrix
      unsigned off = (unsigned)(it * THREADS * 16 + t * 16);
      unsigned lds = (unsigned)(uintptr_t)(bq + off);
      asm volatile("global_load_async_to_lds_b128 %0, %1, off"
                   :: "v"(lds), "v"(gq + off) : "memory");
    }
    unsigned off = (unsigned)((t & 127) * 16);                     // gamma/beta 2 KB each
    const char* gsc = (t < 128) ? ((const char*)lnkv_g + off) : ((const char*)lnkv_b + off);
    unsigned lds = (t < 128) ? (unsigned)(uintptr_t)((char*)gbuf + off)
                             : (unsigned)(uintptr_t)((char*)bbuf + off);
    asm volatile("global_load_async_to_lds_b128 %0, %1, off"
                 :: "v"(lds), "v"(gsc) : "memory");
  }
  asm volatile("s_wait_asynccnt 0" ::: "memory");
  __syncthreads();

  // ---- Phase 2: LayerNorm stats + f32->f16 conversion in place ----
  // wave w owns rows w*8 .. w*8+7; DS ops of a wave are in-order, and store
  // byte-ranges (<=256i+264) never reach future load ranges (>=512(i+1)).
  for (int rr = 0; rr < TILE / 8; ++rr) {
    const int r = wave * (TILE / 8) + rr;
    char* rb = rows + (size_t)r * 2048;
    float4 v[4];
    float sum = 0.f, sq = 0.f;
#pragma unroll
    for (int i = 0; i < 4; ++i) {
      v[i] = *(const float4*)(rb + i * 512 + lane * 16);
      sum += v[i].x + v[i].y + v[i].z + v[i].w;
      sq  += v[i].x * v[i].x + v[i].y * v[i].y + v[i].z * v[i].z + v[i].w * v[i].w;
    }
#pragma unroll
    for (int m = 16; m >= 1; m >>= 1) {
      sum += __shfl_xor(sum, m, 32);
      sq  += __shfl_xor(sq,  m, 32);
    }
    const float mean = sum * (1.f / 512.f);
    const float rstd = rsqrtf(sq * (1.f / 512.f) - mean * mean + LN_EPS);
#pragma unroll
    for (int i = 0; i < 4; ++i) {
      const int e = i * 128 + lane * 4;
      float4 g4 = *(const float4*)(gbuf + e);
      float4 b4 = *(const float4*)(bbuf + e);
      H4 o;
      o.a = (_Float16)((v[i].x - mean) * rstd * g4.x + b4.x);
      o.b = (_Float16)((v[i].y - mean) * rstd * g4.y + b4.y);
      o.c = (_Float16)((v[i].z - mean) * rstd * g4.z + b4.z);
      o.d = (_Float16)((v[i].w - mean) * rstd * g4.w + b4.w);
      *(H4*)(rb + (size_t)e * 2) = o;
    }
  }
  __syncthreads();

  // ---- Phase 3: scores via v_wmma_f32_16x16x32_f16 ----
  // wave -> 16-row subtile (wave&3), K-half (wave>>2); 8 WMMAs each.
  {
    const int st = wave & 3, kh = wave >> 2;
    const int r0 = st * 16;
    const int lh = lane >> 4;
    const char* arow = rows + (size_t)(r0 + (lane & 15)) * 2048;
    const char* bcol = bq + (size_t)(lane & 15) * 1024;
    v8f c = {0.f, 0.f, 0.f, 0.f, 0.f, 0.f, 0.f, 0.f};
#pragma unroll
    for (int ch8 = 0; ch8 < 8; ++ch8) {
      const int ch = kh * 8 + ch8;
      VA a, bm;
      a.q[0]  = *(const int4*)(arow + ch * 64 + lh * 16);        // K 0..7 / 8..15
      a.q[1]  = *(const int4*)(arow + ch * 64 + lh * 16 + 32);   // K 16..23 / 24..31
      bm.q[0] = *(const int4*)(bcol + ch * 64 + lh * 16);        // K 0..7 / 8..15
      bm.q[1] = *(const int4*)(bcol + ch * 64 + lh * 16 + 32);
      c = __builtin_amdgcn_wmma_f32_16x16x32_f16(false, a.h, false, bm.h,
                                                 (short)0, c, false, false);
    }
    float* sc = kh ? scB : scA;
    const int n = lane & 15;              // head column
    if (n < NH) {
#pragma unroll
      for (int vv = 0; vv < 8; ++vv)      // C layout: M = vv + 8*laneHalf
        sc[(r0 + vv + 8 * lh) * NH + n] = c[vv];
    }
  }
  __syncthreads();

  // ---- Phase 4: per-head partial softmax (wave h == head h) ----
  {
    const int h = wave;
    float s0 = scA[lane * NH + h]        + scB[lane * NH + h];
    float s1 = scA[(lane + 32) * NH + h] + scB[(lane + 32) * NH + h];
    float mx = fmaxf(s0, s1);
#pragma unroll
    for (int m = 16; m >= 1; m >>= 1) mx = fmaxf(mx, __shfl_xor(mx, m, 32));
    const float w0 = __expf(s0 - mx), w1 = __expf(s1 - mx);
    float z = w0 + w1;
#pragma unroll
    for (int m = 16; m >= 1; m >>= 1) z += __shfl_xor(z, m, 32);
    wbuf[lane * NH + h] = w0;
    wbuf[(lane + 32) * NH + h] = w1;
    float* Sg = Sout + ((size_t)b * NH + h) * P_SZ + p0;
    Sg[lane] = s0; Sg[lane + 32] = s1;
    if (lane == 0) {
      Mout[((size_t)b * NTILES + tile) * NH + h] = mx;
      Zout[((size_t)b * NTILES + tile) * NH + h] = z;
    }
  }
  __syncthreads();

  // ---- Phase 5: u_partial[h][l] = sum_p w[p][h] * rows_f16[p][l] ----
  {
    const int h = wave;
    float acc[16];
#pragma unroll
    for (int j = 0; j < 16; ++j) acc[j] = 0.f;
    for (int p = 0; p < TILE; ++p) {
      const float wgt = wbuf[p * NH + h];          // LDS broadcast
      VA d;
      d.q[0] = *(const int4*)(rows + (size_t)p * 2048 + lane * 32);
      d.q[1] = *(const int4*)(rows + (size_t)p * 2048 + lane * 32 + 16);
#pragma unroll
      for (int j = 0; j < 16; ++j) acc[j] += wgt * (float)d.h[j];
    }
    float* U = Uout + (((size_t)b * NTILES + tile) * NH + h) * L_SZ + lane * 16;
#pragma unroll
    for (int j = 0; j < 16; ++j) U[j] = acc[j];
  }
}

// ---------------------------------------------------------------------------
// k4: global softmax combine, attn output, ctx = (u @ W_v)/Z + residual
//   grid = B*NH workgroups
// ---------------------------------------------------------------------------
__global__ __launch_bounds__(THREADS)
void k4_combine(const float* __restrict__ Sin, const float* __restrict__ Min,
                const float* __restrict__ Zin, const float* __restrict__ Uin,
                const float* __restrict__ W_v, const float* __restrict__ Rin,
                float* __restrict__ out) {
  __shared__ float u[L_SZ];
  const int bh = blockIdx.x, b = bh >> 3, h = bh & 7;
  const int t = threadIdx.x;
  float mloc[NTILES], esc[NTILES];
  float m = -3.4e38f;
#pragma unroll
  for (int i = 0; i < NTILES; ++i) {
    mloc[i] = Min[((size_t)b * NTILES + i) * NH + h];
    m = fmaxf(m, mloc[i]);
  }
  float Z = 0.f;
#pragma unroll
  for (int i = 0; i < NTILES; ++i) {
    esc[i] = __expf(mloc[i] - m);
    Z += Zin[((size_t)b * NTILES + i) * NH + h] * esc[i];
  }
  const float invZ = 1.f / Z;

  // attention probabilities -> output (after the 32768-float context block)
  const float* S = Sin + ((size_t)b * NH + h) * P_SZ;
  float* attn = out + (size_t)B_SZ * DM + ((size_t)b * NH + h) * P_SZ;
#pragma unroll
  for (int j = 0; j < P_SZ / THREADS; ++j) {
    const int p = t + j * THREADS;
    attn[p] = __expf(S[p] - m) * invZ;
  }

  // combine weighted row sums
  for (int ll = 0; ll < 2; ++ll) {
    const int l = t + ll * 256;
    float acc = 0.f;
    for (int i = 0; i < NTILES; ++i)
      acc += esc[i] * Uin[(((size_t)b * NTILES + i) * NH + h) * L_SZ + l];
    u[l] = acc;
  }
  __syncthreads();

  // ctx[b,h,dk] = (u . W_v[:, h*64+dk]) / Z ; add residual
  const int dk = t >> 2, q = t & 3;
  float acc = 0.f;
  for (int l = q * 128; l < q * 128 + 128; ++l)
    acc += u[l] * W_v[(size_t)l * DM + h * DK + dk];
  acc += __shfl_xor(acc, 1, 32);
  acc += __shfl_xor(acc, 2, 32);
  if (q == 0) {
    const size_t o = (size_t)b * DM + h * DK + dk;
    out[o] = Rin[o] + acc * invZ;
  }
}

// ---------------------------------------------------------------------------
extern "C" void kernel_launch(void* const* d_in, const int* in_sizes, int n_in,
                              void* d_out, int out_size, void* d_ws, size_t ws_size,
                              hipStream_t stream) {
  const float* x_trafic  = (const float*)d_in[0];
  const float* x_dynamic = (const float*)d_in[1];
  // d_in[2] = x_known : unused by the reference computation
  const float* W_q    = (const float*)d_in[3];
  const float* W_k    = (const float*)d_in[4];
  const float* W_v    = (const float*)d_in[5];
  const float* lnq_g  = (const float*)d_in[6];
  const float* lnq_b  = (const float*)d_in[7];
  const float* lnkv_g = (const float*)d_in[8];
  const float* lnkv_b = (const float*)d_in[9];
  const float* res_W  = (const float*)d_in[10];
  const float* res_b  = (const float*)d_in[11];
  float* out = (float*)d_out;
  char*  ws  = (char*)d_ws;

  float*     Q  = (float*)(ws + WS_Q);
  float*     R  = (float*)(ws + WS_RES);
  _Float16*  wq = (_Float16*)(ws + WS_WQ);
  float*     S  = (float*)(ws + WS_S);
  float*     M  = (float*)(ws + WS_M);
  float*     Z  = (float*)(ws + WS_Z);
  float*     U  = (float*)(ws + WS_U);

  k1_query_residual<<<B_SZ, THREADS, 0, stream>>>(x_trafic, W_q, lnq_g, lnq_b,
                                                  res_W, res_b, Q, R);
  k2_wqeff<<<B_SZ, THREADS, 0, stream>>>(W_k, Q, wq);

  const size_t lds3 = (size_t)TILE * 2048 + 16384 + 2048 + 2048 + 3 * TILE * NH * 4;
  k3_attend<<<B_SZ * NTILES, THREADS, lds3, stream>>>(x_dynamic, wq, lnkv_g, lnkv_b,
                                                      S, M, Z, U);
  k4_combine<<<B_SZ * NH, THREADS, 0, stream>>>(S, M, Z, U, W_v, R, out);
}
